// PatchMerging_49100066128646
// MI455X (gfx1250) — compile-verified, matrix operand order checked by reading, not run on hardware
//
#include <hip/hip_runtime.h>
#include <hip/hip_bf16.h>
#include <stddef.h>

// ---- sizes (compile-time, from the reference) ----
#define HH   28
#define WW   28
#define DIMC 384
#define BB   64
#define LTOK 785            // 1 + 28*28
#define NROW 12544          // B * (H/2)*(W/2) = 64*196
#define KTOT 1536           // 4*DIM
#define NOUT 768            // 2*DIM
#define P2   196            // (H/2)*(W/2)
#define ORON 197            // 1 + 196

typedef __bf16 bf16x2  __attribute__((ext_vector_type(2)));
typedef __bf16 bf16x8  __attribute__((ext_vector_type(8)));
typedef __bf16 bf16x16 __attribute__((ext_vector_type(16)));
typedef float  v8f     __attribute__((ext_vector_type(8)));

// -------------------------------------------------------------------------
// Kernel A: patch-merge (2x2) + LayerNorm(1536) + cast to bf16.
// One wave32 per output row. Lane l: quadrant q = l>>3, 48 contiguous
// features at offset (l&7)*48 inside that quadrant -> each lane touches one
// contiguous 192B span of one source pixel row (fully vectorizable loads).
// -------------------------------------------------------------------------
__global__ void __launch_bounds__(256)
merge_ln_bf16(const float* __restrict__ x,
              const float* __restrict__ gamma,
              const float* __restrict__ beta,
              __bf16* __restrict__ mout) {
    const int wave = (int)((blockIdx.x * blockDim.x + threadIdx.x) >> 5);
    const int lane = (int)(threadIdx.x & 31);
    if (wave >= NROW) return;

    const int b  = wave / P2;
    const int p  = wave % P2;
    const int hh = p / 14;
    const int ww = p % 14;

    const int q   = lane >> 3;          // 0..3 : x0,x1,x2,x3
    const int off = (lane & 7) * 48;    // feature offset inside quadrant
    const int h   = 2 * hh + (q & 1);   // x0:(+0,+0) x1:(+1,+0) x2:(+0,+1) x3:(+1,+1)
    const int w   = 2 * ww + (q >> 1);

    const float* __restrict__ src =
        x + ((size_t)b * LTOK + 1 + (size_t)h * WW + w) * DIMC + off;

    float v[48];
    float s = 0.f, ss = 0.f;
#pragma unroll
    for (int i = 0; i < 48; i += 4) {
        float4 t = *(const float4*)(src + i);
        v[i + 0] = t.x; v[i + 1] = t.y; v[i + 2] = t.z; v[i + 3] = t.w;
        s  += t.x + t.y + t.z + t.w;
        ss += t.x * t.x + t.y * t.y + t.z * t.z + t.w * t.w;
    }
    // wave32 butterfly reduction
#pragma unroll
    for (int m = 16; m >= 1; m >>= 1) {
        s  += __shfl_xor(s,  m, 32);
        ss += __shfl_xor(ss, m, 32);
    }
    const float mu   = s * (1.0f / KTOT);
    const float var  = ss * (1.0f / KTOT) - mu * mu;
    const float rstd = rsqrtf(var + 1e-5f);

    const int f0 = q * DIMC + off;
    __bf16* __restrict__ dst = mout + (size_t)wave * KTOT + f0;
#pragma unroll
    for (int i = 0; i < 48; i += 2) {
        float y0 = (v[i + 0] - mu) * rstd * gamma[f0 + i + 0] + beta[f0 + i + 0];
        float y1 = (v[i + 1] - mu) * rstd * gamma[f0 + i + 1] + beta[f0 + i + 1];
        bf16x2 pk; pk.x = (__bf16)y0; pk.y = (__bf16)y1;
        *(bf16x2*)(dst + i) = pk;
    }
}

// -------------------------------------------------------------------------
// Kernel B: fp32 -> bf16 cast of w_reduction (768 x 1536, row-major over f).
// -------------------------------------------------------------------------
__global__ void __launch_bounds__(256)
cast_w_bf16(const float* __restrict__ w, __bf16* __restrict__ wb, int n) {
    int i = (int)(blockIdx.x * blockDim.x + threadIdx.x) * 4;
    if (i >= n) return;
    float4 t = *(const float4*)(w + i);
    bf16x2 p0; p0.x = (__bf16)t.x; p0.y = (__bf16)t.y;
    bf16x2 p1; p1.x = (__bf16)t.z; p1.y = (__bf16)t.w;
    *(bf16x2*)(wb + i)     = p0;
    *(bf16x2*)(wb + i + 2) = p1;
}

// -------------------------------------------------------------------------
// Fragment loader for V_WMMA_F32_16X16X32_BF16.
// 16-bit A/B layout (ISA 7.12.2): lane L holds row (L&15); lanes 0-15 carry
// K = {kk+0..7, kk+16..23}, lanes 16-31 carry K = {kk+8..15, kk+24..31}.
// Both halves are 8 contiguous bf16 = one 16B load each.
// -------------------------------------------------------------------------
__device__ __forceinline__ bf16x16 load_frag(const __bf16* __restrict__ p) {
    bf16x8 lo = *(const bf16x8*)(p);        // K = kbase + 0..7
    bf16x8 hi = *(const bf16x8*)(p + 16);   // K = kbase + 16..23
    return __builtin_shufflevector(lo, hi, 0, 1, 2, 3, 4, 5, 6, 7,
                                           8, 9, 10, 11, 12, 13, 14, 15);
}

// -------------------------------------------------------------------------
// Kernel C: GEMM  out[r, o] = sum_f A[r, f] * W[o, f]   (NT, bf16->f32)
// M=12544, N=768, K=1536. Per-wave tile: 32(M) x 64(N) -> 2x4 accumulators.
// 8 waves per block, each wave owns an independent tile.
// Output row r is remapped to (b*197 + 1 + p) in the final tensor.
// -------------------------------------------------------------------------
__global__ void __launch_bounds__(256)
gemm_wmma_bf16(const __bf16* __restrict__ A,   // NROW x KTOT
               const __bf16* __restrict__ Bw,  // NOUT x KTOT
               float* __restrict__ out) {
    const int lane  = (int)(threadIdx.x & 31);
    const int gwave = (int)((blockIdx.x * blockDim.x + threadIdx.x) >> 5);

    const int NB = NOUT / 64;                // 12 column blocks
    const int mb = gwave / NB;               // 0..391
    const int nb = gwave % NB;               // 0..11
    if (mb >= NROW / 32) return;

    const int row0 = mb * 32;
    const int col0 = nb * 64;

    const int lrow  = lane & 15;
    const int kbase = (lane < 16) ? 0 : 8;

    const __bf16* __restrict__ ap0 = A + (size_t)(row0 + lrow)      * KTOT + kbase;
    const __bf16* __restrict__ ap1 = A + (size_t)(row0 + 16 + lrow) * KTOT + kbase;
    const __bf16* __restrict__ bp0 = Bw + (size_t)(col0 +  0 + lrow) * KTOT + kbase;
    const __bf16* __restrict__ bp1 = Bw + (size_t)(col0 + 16 + lrow) * KTOT + kbase;
    const __bf16* __restrict__ bp2 = Bw + (size_t)(col0 + 32 + lrow) * KTOT + kbase;
    const __bf16* __restrict__ bp3 = Bw + (size_t)(col0 + 48 + lrow) * KTOT + kbase;

    v8f acc[2][4] = {};

#pragma unroll 4
    for (int k = 0; k < KTOT; k += 32) {
        bf16x16 a0 = load_frag(ap0 + k);
        bf16x16 a1 = load_frag(ap1 + k);
        bf16x16 b0 = load_frag(bp0 + k);
        bf16x16 b1 = load_frag(bp1 + k);
        bf16x16 b2 = load_frag(bp2 + k);
        bf16x16 b3 = load_frag(bp3 + k);

        acc[0][0] = __builtin_amdgcn_wmma_f32_16x16x32_bf16(false, a0, false, b0, (short)0, acc[0][0], false, false);
        acc[0][1] = __builtin_amdgcn_wmma_f32_16x16x32_bf16(false, a0, false, b1, (short)0, acc[0][1], false, false);
        acc[0][2] = __builtin_amdgcn_wmma_f32_16x16x32_bf16(false, a0, false, b2, (short)0, acc[0][2], false, false);
        acc[0][3] = __builtin_amdgcn_wmma_f32_16x16x32_bf16(false, a0, false, b3, (short)0, acc[0][3], false, false);
        acc[1][0] = __builtin_amdgcn_wmma_f32_16x16x32_bf16(false, a1, false, b0, (short)0, acc[1][0], false, false);
        acc[1][1] = __builtin_amdgcn_wmma_f32_16x16x32_bf16(false, a1, false, b1, (short)0, acc[1][1], false, false);
        acc[1][2] = __builtin_amdgcn_wmma_f32_16x16x32_bf16(false, a1, false, b2, (short)0, acc[1][2], false, false);
        acc[1][3] = __builtin_amdgcn_wmma_f32_16x16x32_bf16(false, a1, false, b3, (short)0, acc[1][3], false, false);
    }

    // C/D layout: lane l holds col (l&15); VGPR r holds row r + (l<16 ? 0 : 8).
    const int colb = col0 + (lane & 15);
    const int rofs = (lane < 16) ? 0 : 8;
#pragma unroll
    for (int mi = 0; mi < 2; ++mi) {
#pragma unroll
        for (int r = 0; r < 8; ++r) {
            const int grow = row0 + mi * 16 + rofs + r;
            const int ob = grow / P2;
            const int op = grow % P2;
            float* __restrict__ orow = out + ((size_t)ob * ORON + 1 + op) * NOUT;
#pragma unroll
            for (int ni = 0; ni < 4; ++ni) {
                orow[colb + ni * 16] = acc[mi][ni][r];
            }
        }
    }
}

// -------------------------------------------------------------------------
// Kernel D: CLS token path: out[b, 0, o] = sum_c x[b,0,c] * w_convert[o,c]
// Tiny (0.04 GFLOP) -> plain fp32 dot products, full precision.
// -------------------------------------------------------------------------
__global__ void __launch_bounds__(256)
cls_gemm(const float* __restrict__ x, const float* __restrict__ wc,
         float* __restrict__ out) {
    int i = (int)(blockIdx.x * blockDim.x + threadIdx.x);
    if (i >= BB * NOUT) return;
    const int b = i / NOUT;
    const int o = i % NOUT;
    const float* __restrict__ xr = x + (size_t)b * LTOK * DIMC;  // row 0 = CLS
    const float* __restrict__ wr = wc + (size_t)o * DIMC;
    float s = 0.f;
#pragma unroll 4
    for (int c = 0; c < DIMC; c += 4) {
        float4 xa = *(const float4*)(xr + c);
        float4 wa = *(const float4*)(wr + c);
        s += xa.x * wa.x + xa.y * wa.y + xa.z * wa.z + xa.w * wa.w;
    }
    out[(size_t)b * ORON * NOUT + o] = s;
}

// -------------------------------------------------------------------------
extern "C" void kernel_launch(void* const* d_in, const int* in_sizes, int n_in,
                              void* d_out, int out_size, void* d_ws, size_t ws_size,
                              hipStream_t stream) {
    const float* x     = (const float*)d_in[0];   // (64, 785, 384)
    const float* w_red = (const float*)d_in[1];   // (768, 1536)
    const float* w_cvt = (const float*)d_in[2];   // (768, 384)
    const float* gamma = (const float*)d_in[3];   // (1536,)
    const float* beta  = (const float*)d_in[4];   // (1536,)
    float* out = (float*)d_out;                   // (64, 197, 768)

    // workspace: bf16 A (12544x1536) then bf16 W (768x1536)
    __bf16* mA = (__bf16*)d_ws;
    __bf16* wB = (__bf16*)((char*)d_ws + (size_t)NROW * KTOT * sizeof(__bf16));

    // B: cast weights (1,179,648 elems, 4 per thread)
    {
        int n = NOUT * KTOT;
        int thr = n / 4;
        cast_w_bf16<<<(thr + 255) / 256, 256, 0, stream>>>(w_red, wB, n);
    }
    // A: merge + LN + bf16 (one wave per row, 8 waves/block)
    {
        int blocks = (NROW + 7) / 8;
        merge_ln_bf16<<<blocks, 256, 0, stream>>>(x, gamma, beta, mA);
    }
    // C: WMMA GEMM (392*12 = 4704 wave-tiles, 8 per block)
    {
        int waves  = (NROW / 32) * (NOUT / 64);
        int blocks = (waves + 7) / 8;
        gemm_wmma_bf16<<<blocks, 256, 0, stream>>>(mA, wB, out);
    }
    // D: CLS rows
    {
        int n = BB * NOUT;
        cls_gemm<<<(n + 255) / 256, 256, 0, stream>>>(x, w_cvt, out);
    }
}
